// EGNNStack_412316860634
// MI455X (gfx1250) — matrix operand order repaired and checked by reading, not compiled, not run on hardware
//
#include <hip/hip_runtime.h>

// Problem constants (match reference)
#define Nn  100000
#define Ee  600000
#define EPp 100000

typedef _Float16 h16;
typedef __attribute__((ext_vector_type(16))) _Float16 v16h;
typedef __attribute__((ext_vector_type(2)))  _Float16 v2h;
typedef __attribute__((ext_vector_type(8)))  float    v8f;
typedef __attribute__((ext_vector_type(4)))  unsigned int u32x4;
typedef __attribute__((ext_vector_type(8)))  int          i32x8;
typedef __attribute__((ext_vector_type(4)))  int          i32x4;

__device__ __forceinline__ void atomAddF(float* p, float v) {
    (void)__hip_atomic_fetch_add(p, v, __ATOMIC_RELAXED, __HIP_MEMORY_SCOPE_AGENT);
}

// ---------------- graph norm kernels ----------------
__global__ __launch_bounds__(256) void k_init_deg(float* deg, int n) {
    int i = blockIdx.x * 256 + threadIdx.x;
    if (i < n) deg[i] = 1.0f;                      // self-loop contributes 1
}
__global__ __launch_bounds__(256) void k_deg_edges(const int* __restrict__ dst,
                                                   float* __restrict__ deg, int e) {
    int i = blockIdx.x * 256 + threadIdx.x;
    if (i < e) atomAddF(&deg[dst[i]], 1.0f);
}
__global__ __launch_bounds__(256) void k_dinv(float* deg, int n) {
    int i = blockIdx.x * 256 + threadIdx.x;
    if (i < n) { float d = deg[i]; deg[i] = (d > 0.0f) ? rsqrtf(d) : 0.0f; }
}
__global__ __launch_bounds__(256) void k_norm(const int* __restrict__ src,
                                              const int* __restrict__ dst,
                                              const float* __restrict__ dinv,
                                              float* __restrict__ norm, int e) {
    int i = blockIdx.x * 256 + threadIdx.x;
    if (i < e) norm[i] = dinv[src[i]] * dinv[dst[i]];
}

// ---------------- precision / layout prep ----------------
__global__ __launch_bounds__(256) void k_cvt_f16(const float* __restrict__ in,
                                                 h16* __restrict__ out, int n) {
    int i = blockIdx.x * 256 + threadIdx.x;
    if (i < n) out[i] = (h16)in[i];
}

// Re-pack a [K,128] f32 weight into the per-lane WMMA B-fragment layout:
// element (k,c) -> out[((kc*2 + h)*128 + c)*16 + j], kc=k/32, h=(k%32)/16, j=k%16.
// A lane (col c, khalf h) then reads one contiguous 32B run per K-chunk.
__global__ __launch_bounds__(256) void k_prep_B(const float* __restrict__ in,
                                                h16* __restrict__ out, int K) {
    int tid = blockIdx.x * 256 + threadIdx.x;
    if (tid >= K * 128) return;
    int k = tid >> 7, c = tid & 127;
    int kc = k >> 5, r = k & 31, h = r >> 4, j = r & 15;
    out[(((size_t)((kc << 1) + h) << 7) + c) * 16 + j] = (h16)in[tid];
}

// ---------------- WMMA GEMM: [M,K]@[K,128] ----------------
// Block: 32 rows x 128 cols, 256 threads = 8 waves; wave w -> cols 16w..16w+15,
// two 16x16 accumulators (rows 0-15 and 16-31) sharing one B fragment.
// mode 0: A rows streamed into LDS by the Tensor Data Mover (TDM).
// mode 1: A row = [xsrc[gsrc[r]] | xsrc[gdst[r]]], K=256, cooperative gather fill.
__global__ __launch_bounds__(256) void k_gemm(
    const h16* __restrict__ A, const h16* __restrict__ xsrc,
    const int* __restrict__ gsrc, const int* __restrict__ gdst,
    const h16* __restrict__ Bp, const float* __restrict__ bias,
    float* __restrict__ outf, h16* __restrict__ outh,
    int K, int mode, int relu)
{
    extern __shared__ h16 Atile[];                 // [32][K] f16
    const int row0 = blockIdx.x * 32;
    const int wave = threadIdx.x >> 5;

    if (mode == 0) {
        // ---- TDM: DMA the dense 32xK f16 tile straight into LDS ----
        if (wave == 0) {
            unsigned lds_off = (unsigned)(unsigned long long)(void*)Atile; // flat[31:0] == LDS offset
            unsigned long long ga = (unsigned long long)(const void*)(A + (size_t)row0 * K);
            u32x4 g0;
            g0[0] = 1u;                                            // count=1, no gather
            g0[1] = lds_off;                                       // lds_addr
            g0[2] = (unsigned)(ga & 0xFFFFFFFFu);                  // global_addr[31:0]
            g0[3] = (unsigned)((ga >> 32) & 0x1FFFFFFu) | (2u << 30); // addr[56:32] | type=2
            const unsigned dim1 = 1u << 20;                        // generous row bound (never OOB)
            i32x8 g1;
            g1[0] = (int)(1u << 16);                               // wg_mask=0, data_size=2B
            g1[1] = (int)(((unsigned)K & 0xFFFFu) << 16);          // tensor_dim0[15:0]
            g1[2] = (int)((((unsigned)K >> 16) & 0xFFFFu)          // tensor_dim0[31:16]
                          | ((dim1 & 0xFFFFu) << 16));             // tensor_dim1[15:0]
            g1[3] = (int)(((dim1 >> 16) & 0xFFFFu)                 // tensor_dim1[31:16]
                          | (((unsigned)K & 0xFFFFu) << 16));      // tile_dim0 = K
            g1[4] = 32;                                            // tile_dim1 = 32, tile_dim2 = 0
            g1[5] = K;                                             // tensor_dim0_stride[31:0]
            g1[6] = 0;                                             // stride hi, dim1_stride lo
            g1[7] = 0;
            i32x4 gz4; gz4[0] = 0; gz4[1] = 0; gz4[2] = 0; gz4[3] = 0; // groups 2/3 unused (2D tile)
            i32x8 gz8;
            gz8[0] = 0; gz8[1] = 0; gz8[2] = 0; gz8[3] = 0;
            gz8[4] = 0; gz8[5] = 0; gz8[6] = 0; gz8[7] = 0;
            __builtin_amdgcn_tensor_load_to_lds(g0, g1, gz4, gz4, gz8, 0);
            __builtin_amdgcn_s_wait_tensorcnt(0);
        }
    } else {
        // ---- cooperative indirect fill: row = [xsrc[gsrc[r]] | xsrc[gdst[r]]] ----
        const int K8 = K >> 3;
        for (int v = threadIdx.x; v < 32 * K8; v += 256) {
            int r = v / K8;
            int c = (v - r * K8) << 3;             // multiple of 8 -> 16B aligned
            int grow = row0 + r;
            int node = (c < 128) ? gsrc[grow] : gdst[grow];
            const h16* sp = xsrc + ((size_t)node << 7) + (c & 127);
            *(uint4*)(Atile + r * K + c) = *(const uint4*)sp;
        }
    }
    __syncthreads();

    const int lane  = threadIdx.x & 31;
    const int l15   = lane & 15;
    const int khalf = lane >> 4;
    const int colg  = (wave << 4) + l15;

    v8f acc0 = {}; v8f acc1 = {};
    const int nchunk = K >> 5;
    for (int kc = 0; kc < nchunk; ++kc) {
        const int kb = kc << 5;
        // contiguous 32B B-fragment load (pre-swizzled)
        v16h bfrag = *(const v16h*)(Bp + (((size_t)(((kc << 1) + khalf) << 7) + colg) << 4));
        v16h a0, a1;
#pragma unroll
        for (int v = 0; v < 8; ++v) {
            // contiguous A layout matching B fragment: K=2v+{0,1}; +16*khalf
            int kp = kb + (v << 1) + (khalf << 4);
            v2h p0 = *(const v2h*)(Atile + l15 * K + kp);
            v2h p1 = *(const v2h*)(Atile + (l15 + 16) * K + kp);
            a0[2 * v] = p0[0]; a0[2 * v + 1] = p0[1];
            a1[2 * v] = p1[0]; a1[2 * v + 1] = p1[1];
        }
        acc0 = __builtin_amdgcn_wmma_f32_16x16x32_f16(false, a0, false, bfrag,
                                                      (short)0, acc0, false, false);
        acc1 = __builtin_amdgcn_wmma_f32_16x16x32_f16(false, a1, false, bfrag,
                                                      (short)0, acc1, false, false);
    }

    const float bv = bias ? bias[colg] : 0.0f;
#pragma unroll
    for (int i = 0; i < 8; ++i) {
        int m = i + (khalf << 3);                  // D layout: VGPR i -> M=i / M=i+8
        float v0 = acc0[i] + bv;
        float v1 = acc1[i] + bv;
        if (relu) { v0 = fmaxf(v0, 0.0f); v1 = fmaxf(v1, 0.0f); }
        size_t o0 = (((size_t)(row0 + m)) << 7) + colg;
        size_t o1 = (((size_t)(row0 + 16 + m)) << 7) + colg;
        if (outf) { outf[o0] = v0;       outf[o1] = v1; }
        else      { outh[o0] = (h16)v0;  outh[o1] = (h16)v1; }
    }
}

// ---------------- GCN aggregation ----------------
__global__ __launch_bounds__(256) void k_agg_self(const float* __restrict__ h,
                                                  const float* __restrict__ dinv,
                                                  float* __restrict__ agg, int total) {
    int g = blockIdx.x * 256 + threadIdx.x;
    if (g >= total) return;
    float di = dinv[g >> 7];
    agg[g] = di * di * h[g];
}
__global__ __launch_bounds__(256) void k_agg_edges(const float* __restrict__ h,
                                                   const float* __restrict__ norm,
                                                   const int* __restrict__ src,
                                                   const int* __restrict__ dst,
                                                   float* __restrict__ agg, int total) {
    int g = blockIdx.x * 256 + threadIdx.x;
    if (g >= total) return;
    int e = g >> 7, f = g & 127;
    float v = norm[e] * h[((size_t)src[e] << 7) + f];
    atomAddF(&agg[((size_t)dst[e] << 7) + f], v);
}
__global__ __launch_bounds__(256) void k_finish_node(const float* __restrict__ agg,
                                                     const float* __restrict__ bias,
                                                     h16* __restrict__ xout, int total) {
    int g = blockIdx.x * 256 + threadIdx.x;
    if (g >= total) return;
    xout[g] = (h16)fmaxf(agg[g] + bias[g & 127], 0.0f);
}

// ---------------- final H->1 layer ----------------
__global__ __launch_bounds__(256) void k_final(const h16* __restrict__ t,
                                               const float* __restrict__ w,
                                               const float* __restrict__ b,
                                               float* __restrict__ y, int M) {
    int i = blockIdx.x * 256 + threadIdx.x;
    if (i >= M) return;
    const h16* row = t + ((size_t)i << 7);
    float s = b[0];
#pragma unroll
    for (int f = 0; f < 128; f += 2) {
        v2h p = *(const v2h*)(row + f);
        s += (float)p[0] * w[f] + (float)p[1] * w[f + 1];
    }
    y[i] = fmaxf(s, 0.0f);
}

// ---------------- host ----------------
static inline char* carve(char*& p, size_t bytes) {
    char* r = p;
    p += (bytes + 255) & ~(size_t)255;
    return r;
}

extern "C" void kernel_launch(void* const* d_in, const int* in_sizes, int n_in,
                              void* d_out, int out_size, void* d_ws, size_t ws_size,
                              hipStream_t stream) {
    (void)in_sizes; (void)n_in; (void)out_size; (void)ws_size;

    const float* x    = (const float*)d_in[0];
    // d_in[1] edge_attr: dead (edge MLPs do not affect the returned y)
    const int*   ei   = (const int*)d_in[2];   // [2,E]
    const int*   pei  = (const int*)d_in[3];   // [2,EP]
    const float* W1   = (const float*)d_in[4];
    const float* b1   = (const float*)d_in[5];
    const float* W2   = (const float*)d_in[6];
    const float* b2   = (const float*)d_in[7];
    const float* pW1  = (const float*)d_in[12];
    const float* pb1  = (const float*)d_in[13];
    const float* pW2  = (const float*)d_in[14];
    const float* pb2  = (const float*)d_in[15];
    const float* pW3  = (const float*)d_in[16];
    const float* pb3  = (const float*)d_in[17];
    const float* pW4  = (const float*)d_in[18];
    const float* pb4  = (const float*)d_in[19];
    float* y = (float*)d_out;

    const int* esrc = ei;          const int* edst = ei + Ee;
    const int* psrc = pei;         const int* pdst = pei + EPp;

    char* wp = (char*)d_ws;
    float* dinv = (float*)carve(wp, (size_t)Nn * 4);
    float* norm = (float*)carve(wp, (size_t)Ee * 4);
    h16*   xb   = (h16*)  carve(wp, (size_t)Nn * 128 * 2);
    h16*   x1b  = (h16*)  carve(wp, (size_t)Nn * 128 * 2);
    h16*   x2b  = (h16*)  carve(wp, (size_t)Nn * 128 * 2);
    float* hbuf = (float*)carve(wp, (size_t)Nn * 128 * 4);
    float* agg  = (float*)carve(wp, (size_t)Nn * 128 * 4);
    h16*   t1   = (h16*)  carve(wp, (size_t)EPp * 128 * 2);
    h16*   t2   = (h16*)  carve(wp, (size_t)EPp * 128 * 2);
    h16*   W1p  = (h16*)  carve(wp, (size_t)128 * 128 * 2);
    h16*   W2p  = (h16*)  carve(wp, (size_t)128 * 128 * 2);
    h16*   pW1p = (h16*)  carve(wp, (size_t)256 * 128 * 2);
    h16*   pW2p = (h16*)  carve(wp, (size_t)128 * 128 * 2);
    h16*   pW3p = (h16*)  carve(wp, (size_t)128 * 128 * 2);

    const int NF = Nn * 128;               // 12.8M
    const int EF = Ee * 128;               // 76.8M
    const int gN   = (Nn + 255) / 256;
    const int gE   = (Ee + 255) / 256;
    const int gNF  = NF / 256;
    const int gEF  = EF / 256;
    const int gemmN  = Nn / 32;            // 3125 (exact)
    const int gemmEP = EPp / 32;           // 3125 (exact)
    const size_t lds128 = 32 * 128 * 2;
    const size_t lds256 = 32 * 256 * 2;

    // graph normalization
    k_init_deg <<<gN, 256, 0, stream>>>(dinv, Nn);
    k_deg_edges<<<gE, 256, 0, stream>>>(edst, dinv, Ee);
    k_dinv     <<<gN, 256, 0, stream>>>(dinv, Nn);
    k_norm     <<<gE, 256, 0, stream>>>(esrc, edst, dinv, norm, Ee);

    // precision + weight-fragment prep
    k_cvt_f16<<<gNF, 256, 0, stream>>>(x, xb, NF);
    k_prep_B <<<(128 * 128) / 256, 256, 0, stream>>>(W1,  W1p,  128);
    k_prep_B <<<(128 * 128) / 256, 256, 0, stream>>>(W2,  W2p,  128);
    k_prep_B <<<(256 * 128) / 256, 256, 0, stream>>>(pW1, pW1p, 256);
    k_prep_B <<<(128 * 128) / 256, 256, 0, stream>>>(pW2, pW2p, 128);
    k_prep_B <<<(128 * 128) / 256, 256, 0, stream>>>(pW3, pW3p, 128);

    // GCN layer 0: h = x@W1 ; agg ; relu(agg + b1) -> x1b (f16)
    k_gemm<<<gemmN, 256, lds128, stream>>>(xb, nullptr, nullptr, nullptr, W1p,
                                           nullptr, hbuf, nullptr, 128, 0, 0);
    k_agg_self  <<<gNF, 256, 0, stream>>>(hbuf, dinv, agg, NF);
    k_agg_edges <<<gEF, 256, 0, stream>>>(hbuf, norm, esrc, edst, agg, EF);
    k_finish_node<<<gNF, 256, 0, stream>>>(agg, b1, x1b, NF);

    // GCN layer 1: h = x1@W2 ; agg ; relu(agg + b2) -> x2b
    k_gemm<<<gemmN, 256, lds128, stream>>>(x1b, nullptr, nullptr, nullptr, W2p,
                                           nullptr, hbuf, nullptr, 128, 0, 0);
    k_agg_self  <<<gNF, 256, 0, stream>>>(hbuf, dinv, agg, NF);
    k_agg_edges <<<gEF, 256, 0, stream>>>(hbuf, norm, esrc, edst, agg, EF);
    k_finish_node<<<gNF, 256, 0, stream>>>(agg, b2, x2b, NF);

    // predict head: fused [x_i|x_j] gather-GEMM, then two 128x128 layers, then dot
    k_gemm<<<gemmEP, 256, lds256, stream>>>(nullptr, x2b, psrc, pdst, pW1p,
                                            pb1, nullptr, t1, 256, 1, 1);
    k_gemm<<<gemmEP, 256, lds128, stream>>>(t1, nullptr, nullptr, nullptr, pW2p,
                                            pb2, nullptr, t2, 128, 0, 1);
    k_gemm<<<gemmEP, 256, lds128, stream>>>(t2, nullptr, nullptr, nullptr, pW3p,
                                            pb3, nullptr, t1, 128, 0, 1);
    k_final<<<(EPp + 255) / 256, 256, 0, stream>>>(t1, pW4, pb4, y, EPp);
}
